// MambaClassifier_6682969113245
// MI455X (gfx1250) — matrix operand order repaired and checked by reading, not compiled, not run on hardware
//
#include <hip/hip_runtime.h>
#include <hip/hip_bf16.h>
#include <math.h>

// ---------------- model constants ----------------
#define N_LAYERS 2
#define D_MODEL  512
#define D_STATE  16
#define D_CONV   4
#define D_INNER  1024          // 2 * D_MODEL
#define DT_RANK  32            // ceil(512/16)
#define BATCH    16
#define SEQ      2048
#define MROWS    (BATCH * SEQ) // 32768

typedef __bf16 bhalf;
typedef bhalf v16bh __attribute__((ext_vector_type(16)));
typedef bhalf v8bh  __attribute__((ext_vector_type(8)));
typedef float v8f   __attribute__((ext_vector_type(8)));
typedef int   v4i   __attribute__((ext_vector_type(4)));

#if __has_builtin(__builtin_amdgcn_global_load_async_to_lds_b128) && \
    __has_builtin(__builtin_amdgcn_s_wait_asynccnt)
#define HAVE_ASYNC_LDS 1
#else
#define HAVE_ASYNC_LDS 0
#endif

// ---------------- fp32 -> bf16 conversion ----------------
__global__ void k_f32_to_bf16(const float* __restrict__ in, bhalf* __restrict__ out, int n) {
    int i = blockIdx.x * blockDim.x + threadIdx.x;
    if (i < n) out[i] = (bhalf)in[i];
}

// ---------------- RMSNorm: one block per (b,l) row; bf16 output ----------------
__global__ void k_rmsnorm(const float* __restrict__ x, const float* __restrict__ w,
                          bhalf* __restrict__ y) {
    int row = blockIdx.x;
    const float* xr = x + (size_t)row * D_MODEL;
    bhalf*       yr = y + (size_t)row * D_MODEL;
    int t = threadIdx.x;                     // 256 threads, 2 cols each
    float a0 = xr[t], a1 = xr[t + 256];
    __shared__ float red[256];
    red[t] = a0 * a0 + a1 * a1;
    __syncthreads();
    for (int s = 128; s > 0; s >>= 1) {
        if (t < s) red[t] += red[t + s];
        __syncthreads();
    }
    float scale = rsqrtf(red[0] / (float)D_MODEL + 1e-5f);
    yr[t]       = (bhalf)(a0 * scale * w[t]);
    yr[t + 256] = (bhalf)(a1 * scale * w[t + 256]);
}

// ---------------- A-fragment loaders (16x32, CDNA5 ISA layout) ----------------
// lane row = lane&15; lanes 0-15 hold K=[k..k+7],[k+16..k+23]; lanes 16-31 hold [k+8..],[k+24..]
__device__ __forceinline__ v16bh load_a_frag(const float* __restrict__ A, int lda,
                                             int row, int k, int lane) {
    const float* p = A + (size_t)row * lda + k + ((lane >> 4) << 3);
    float4 f0 = *(const float4*)(p);
    float4 f1 = *(const float4*)(p + 4);
    float4 f2 = *(const float4*)(p + 16);
    float4 f3 = *(const float4*)(p + 20);
    v16bh a;
    a[0]  = (bhalf)f0.x; a[1]  = (bhalf)f0.y; a[2]  = (bhalf)f0.z; a[3]  = (bhalf)f0.w;
    a[4]  = (bhalf)f1.x; a[5]  = (bhalf)f1.y; a[6]  = (bhalf)f1.z; a[7]  = (bhalf)f1.w;
    a[8]  = (bhalf)f2.x; a[9]  = (bhalf)f2.y; a[10] = (bhalf)f2.z; a[11] = (bhalf)f2.w;
    a[12] = (bhalf)f3.x; a[13] = (bhalf)f3.y; a[14] = (bhalf)f3.z; a[15] = (bhalf)f3.w;
    return a;
}
__device__ __forceinline__ v16bh load_a_frag(const bhalf* __restrict__ A, int lda,
                                             int row, int k, int lane) {
    const bhalf* p = A + (size_t)row * lda + k + ((lane >> 4) << 3);
    v8bh lo = *(const v8bh*)(p);
    v8bh hi = *(const v8bh*)(p + 16);
    v16bh a;
#pragma unroll
    for (int i = 0; i < 8; ++i) { a[i] = lo[i]; a[8 + i] = hi[i]; }
    return a;
}

// ---------------- stage one 32-K slice of B (ROWS x 32 bf16) into LDS ----------------
// Row c is 64 bytes contiguous in W; each thread moves its row as 4 x b128 lane-transfers.
__device__ __forceinline__ void stage_b(const bhalf* __restrict__ W, int K, int n0, int k,
                                        bhalf* lds, int rows, int t) {
    if (t < rows) {
        const bhalf* g = W + (size_t)(n0 + t) * K + k;   // 64B, 16B-aligned
        bhalf* l = lds + t * 32;
#if HAVE_ASYNC_LDS
#pragma unroll
        for (int j = 0; j < 4; ++j) {
            void* gp = const_cast<bhalf*>(g + j * 8);
            void* lp = (void*)(l + j * 8);
            __builtin_amdgcn_global_load_async_to_lds_b128(
                (__attribute__((address_space(1))) v4i*)gp,
                (__attribute__((address_space(3))) v4i*)lp, 0, 0);
        }
#else
        const uint4* gs = (const uint4*)g;
        uint4*       ls = (uint4*)l;
#pragma unroll
        for (int j = 0; j < 4; ++j) ls[j] = gs[j];
#endif
    }
}

// ---------------- WMMA bf16 GEMM: C[M,N] = A[M,K] * W[N,K]^T ----------------
// Block = 4 waves stacked in M (64 rows), sharing a (NSUB*16)-column B slab staged in LDS
// with double buffering. Each wave owns a 16 x (NSUB*16) C tile (NSUB WMMAs per k-step).
// Requires M%64==0, N%(NSUB*16)==0, K%32==0.
template <typename AT, int NSUB>
__global__ void k_gemm_wmma(const AT* __restrict__ A, int lda,
                            const bhalf* __restrict__ W,   // N x K row-major (bf16)
                            float* __restrict__ C, int ldc,
                            int M, int N, int K,
                            const float* __restrict__ bias, int softplus_en,
                            const float* __restrict__ res) {
    constexpr int ROWS = NSUB * 16;
    __shared__ bhalf ldsB[2][ROWS * 32];
    int t    = threadIdx.x;              // 128 threads = 4 waves
    int wave = t >> 5;
    int lane = t & 31;
    int mBlks = M >> 6;
    int mb = blockIdx.x % mBlks;
    int nb = blockIdx.x / mBlks;
    int m0 = (mb * 4 + wave) << 4;
    int n0 = nb * ROWS;
    int rowA = m0 + (lane & 15);

    v8f acc[NSUB];
#pragma unroll
    for (int s = 0; s < NSUB; ++s) acc[s] = (v8f){};

#if HAVE_ASYNC_LDS
    stage_b(W, K, n0, 0, &ldsB[0][0], ROWS, t);
#endif
    for (int k = 0; k < K; k += 32) {
        int cur = (k >> 5) & 1;
#if HAVE_ASYNC_LDS
        if (k + 32 < K) {
            stage_b(W, K, n0, k + 32, &ldsB[cur ^ 1][0], ROWS, t);
            __builtin_amdgcn_s_wait_asynccnt(4);   // previous slice complete (in-order)
        } else {
            __builtin_amdgcn_s_wait_asynccnt(0);
        }
        __syncthreads();
#else
        stage_b(W, K, n0, k, &ldsB[cur][0], ROWS, t);
        __syncthreads();
#endif
        if (k + 64 < K)   // prefetch A stream two k-steps ahead -> global_prefetch_b8
            __builtin_prefetch(A + (size_t)rowA * lda + k + 64, 0, 0);

        v16bh a = load_a_frag(A, lda, rowA, k, lane);
        // B fragment (32x16): lane col = lane&15; lanes 0-15 hold K=[0..7],[16..23]; lanes 16-31 hold [8..15],[24..31]
#pragma unroll
        for (int s = 0; s < NSUB; ++s) {
            const bhalf* lb = &ldsB[cur][(s * 16 + (lane & 15)) * 32 + ((lane >> 4) << 3)];
            v8bh blo = *(const v8bh*)lb;
            v8bh bhi = *(const v8bh*)(lb + 16);
            v16bh bs;
#pragma unroll
            for (int i = 0; i < 8; ++i) { bs[i] = blo[i]; bs[8 + i] = bhi[i]; }
            acc[s] = __builtin_amdgcn_wmma_f32_16x16x32_bf16(
                false, a, false, bs, (short)0, acc[s], false, false);
        }
        __syncthreads();   // all reads of this buffer done before it is restaged
    }

    // C layout: VGPR r -> M = m0 + r + 8*(lane>=16), N = n0 + s*16 + (lane&15)
    int rowC = m0 + ((lane >> 4) << 3);
    int colB = n0 + (lane & 15);
#pragma unroll
    for (int r = 0; r < 8; ++r) {
        int row = rowC + r;
#pragma unroll
        for (int s = 0; s < NSUB; ++s) {
            int col = colB + s * 16;
            float o = acc[s][r];
            if (bias)        o += bias[col];
            if (softplus_en) o = (o > 20.f) ? o : log1pf(__expf(o));
            if (res)         o += res[(size_t)row * ldc + col];
            C[(size_t)row * ldc + col] = o;
        }
    }
}

// ---------------- causal depthwise conv (k=4) + bias + SiLU ----------------
__global__ void k_conv_silu(const float* __restrict__ xz,       // (M, 2*D_INNER), xb = cols [0,D_INNER)
                            const float* __restrict__ convW,    // (D_INNER, D_CONV)
                            const float* __restrict__ convB,    // (D_INNER)
                            float* __restrict__ xc) {           // (M, D_INNER)
    int idx = blockIdx.x * blockDim.x + threadIdx.x;
    if (idx >= MROWS * D_INNER) return;
    int e = idx & (D_INNER - 1);
    int t = (idx >> 10) & (SEQ - 1);
    int b = idx >> 21;
    size_t base = (size_t)b * SEQ;
    float acc = convB[e];
#pragma unroll
    for (int j = 0; j < D_CONV; ++j) {
        int tt = t - (D_CONV - 1) + j;
        if (tt >= 0)
            acc += xz[(base + tt) * (2 * D_INNER) + e] * convW[e * D_CONV + j];
    }
    acc = acc / (1.f + __expf(-acc));      // SiLU
    xc[idx] = acc;
}

// ---------------- selective scan, fused D-skip + SiLU(z) gate; bf16 y out ----------------
__global__ void k_ssm_scan(const float* __restrict__ delta,   // (M, D_INNER)
                           const float* __restrict__ xc,      // (M, D_INNER)
                           const float* __restrict__ dbc,     // (M, 64) = [dt(32)|B(16)|C(16)]
                           const float* __restrict__ xz,      // z at col offset D_INNER
                           const float* __restrict__ A_log,   // (D_INNER, D_STATE)
                           const float* __restrict__ Dskip,   // (D_INNER)
                           bhalf* __restrict__ y) {           // (M, D_INNER) bf16
    int tid = blockIdx.x * blockDim.x + threadIdx.x;   // 16384 threads
    int e = tid & (D_INNER - 1);
    int b = tid >> 10;
    float a[D_STATE], h[D_STATE];
#pragma unroll
    for (int n = 0; n < D_STATE; ++n) {
        a[n] = -__expf(A_log[e * D_STATE + n]);
        h[n] = 0.f;
    }
    float dsk = Dskip[e];
    size_t rowBase = (size_t)b * SEQ;
    for (int t = 0; t < SEQ; ++t) {
        size_t row = rowBase + t;
        float d = delta[row * D_INNER + e];
        float x = xc[row * D_INNER + e];
        const float4* bc = (const float4*)(dbc + row * 64 + DT_RANK);   // wave-uniform address
        float4 B0 = bc[0], B1 = bc[1], B2 = bc[2], B3 = bc[3];
        float4 C0 = bc[4], C1 = bc[5], C2 = bc[6], C3 = bc[7];
        float Bv[16] = { B0.x,B0.y,B0.z,B0.w, B1.x,B1.y,B1.z,B1.w,
                         B2.x,B2.y,B2.z,B2.w, B3.x,B3.y,B3.z,B3.w };
        float Cv[16] = { C0.x,C0.y,C0.z,C0.w, C1.x,C1.y,C1.z,C1.w,
                         C2.x,C2.y,C2.z,C2.w, C3.x,C3.y,C3.z,C3.w };
        float dx = d * x;
        float yv = 0.f;
#pragma unroll
        for (int n = 0; n < D_STATE; ++n) {
            h[n] = __expf(d * a[n]) * h[n] + dx * Bv[n];
            yv += h[n] * Cv[n];
        }
        float z  = xz[row * (2 * D_INNER) + D_INNER + e];
        float sz = z / (1.f + __expf(-z));
        y[row * D_INNER + e] = (bhalf)((yv + dsk * x) * sz);
    }
}

// ---------------- mean-pool over L + classifier + sigmoid ----------------
__global__ void k_pool_cls(const float* __restrict__ x,      // (B, SEQ, D_MODEL)
                           const float* __restrict__ clsW,   // (1, D_MODEL)
                           const float* __restrict__ clsb,   // (1)
                           float* __restrict__ out) {        // (B, 1)
    int b = blockIdx.x;
    int d = threadIdx.x;                    // 512 threads
    const float* xb = x + (size_t)b * SEQ * D_MODEL;
    float s = 0.f;
    for (int t = 0; t < SEQ; ++t) s += xb[(size_t)t * D_MODEL + d];
    __shared__ float red[512];
    red[d] = (s / (float)SEQ) * clsW[d];
    __syncthreads();
    for (int st = 256; st > 0; st >>= 1) {
        if (d < st) red[d] += red[d + st];
        __syncthreads();
    }
    if (d == 0) {
        float logit = red[0] + clsb[0];
        out[b] = 1.f / (1.f + __expf(-logit));
    }
}

// ---------------- host launcher ----------------
extern "C" void kernel_launch(void* const* d_in, const int* in_sizes, int n_in,
                              void* d_out, int out_size, void* d_ws, size_t ws_size,
                              hipStream_t stream) {
    (void)in_sizes; (void)n_in; (void)out_size; (void)ws_size;
    const float* src    = (const float*)d_in[0];
    const float* norm_w = (const float*)d_in[1];
    const float* inW    = (const float*)d_in[2];
    const float* convW  = (const float*)d_in[3];
    const float* convB  = (const float*)d_in[4];
    const float* xW     = (const float*)d_in[5];
    const float* dtW    = (const float*)d_in[6];
    const float* dtB    = (const float*)d_in[7];
    const float* A_log  = (const float*)d_in[8];
    const float* Dskip  = (const float*)d_in[9];
    const float* outW   = (const float*)d_in[10];
    const float* clsW   = (const float*)d_in[11];
    const float* clsB   = (const float*)d_in[12];
    float* out = (float*)d_out;

    // workspace carve-out
    char* ws = (char*)d_ws;
    size_t off = 0;
    auto alloc = [&](size_t bytes) -> char* {
        char* p = ws + off;
        off += (bytes + 255) & ~(size_t)255;
        return p;
    };
    const size_t M = (size_t)MROWS;
    bhalf* xn   = (bhalf*)alloc(M * D_MODEL * 2);          // rmsnorm output (bf16)
    float* xzb  = (float*)alloc(M * 2 * D_INNER * 4);      // in_proj output (xb|z)
    float* xcb  = (float*)alloc(M * D_INNER * 4);          // conv+silu output
    float* dbcb = (float*)alloc(M * 64 * 4);               // x_proj output (dt|B|C)
    float* dly  = (float*)alloc(M * D_INNER * 4);          // delta (softplus output)
    bhalf* ybf  = (bhalf*)alloc(M * D_INNER * 2);          // scan output y (bf16)
    float* x0   = (float*)alloc(M * D_MODEL * 4);          // layer0 residual output
    float* x1   = (float*)alloc(M * D_MODEL * 4);          // layer1 residual output
    bhalf* inWb  = (bhalf*)alloc((size_t)N_LAYERS * 2 * D_INNER * D_MODEL * 2);
    bhalf* xWb   = (bhalf*)alloc((size_t)N_LAYERS * 64 * D_INNER * 2);
    bhalf* dtWb  = (bhalf*)alloc((size_t)N_LAYERS * D_INNER * DT_RANK * 2);
    bhalf* outWb = (bhalf*)alloc((size_t)N_LAYERS * D_MODEL * D_INNER * 2);

    auto cvt = [&](const float* s, bhalf* dptr, size_t n) {
        k_f32_to_bf16<<<(unsigned)((n + 255) / 256), 256, 0, stream>>>(s, dptr, (int)n);
    };
    cvt(inW,  inWb,  (size_t)N_LAYERS * 2 * D_INNER * D_MODEL);
    cvt(xW,   xWb,   (size_t)N_LAYERS * 64 * D_INNER);
    cvt(dtW,  dtWb,  (size_t)N_LAYERS * D_INNER * DT_RANK);
    cvt(outW, outWb, (size_t)N_LAYERS * D_MODEL * D_INNER);

    const float* xin = src;
    for (int l = 0; l < N_LAYERS; ++l) {
        float* xout = (l == 0) ? x0 : x1;
        // 1) RMSNorm -> bf16
        k_rmsnorm<<<(int)M, 256, 0, stream>>>(xin, norm_w + l * D_MODEL, xn);
        // 2) in_proj: (M,512)bf16 x (2048,512)^T -> xz (M,2048)   [NSUB=8 -> 128-col slab]
        k_gemm_wmma<bhalf, 8><<<(MROWS / 64) * (2 * D_INNER / 128), 128, 0, stream>>>(
            xn, D_MODEL, inWb + (size_t)l * 2 * D_INNER * D_MODEL,
            xzb, 2 * D_INNER, MROWS, 2 * D_INNER, D_MODEL, nullptr, 0, nullptr);
        // 3) causal depthwise conv + bias + SiLU on xb
        k_conv_silu<<<(MROWS * D_INNER + 255) / 256, 256, 0, stream>>>(
            xzb, convW + (size_t)l * D_INNER * D_CONV, convB + l * D_INNER, xcb);
        // 4) x_proj: (M,1024)f32 x (64,1024)^T -> dbc (M,64)      [NSUB=4 -> 64-col slab]
        k_gemm_wmma<float, 4><<<(MROWS / 64) * (64 / 64), 128, 0, stream>>>(
            xcb, D_INNER, xWb + (size_t)l * 64 * D_INNER,
            dbcb, 64, MROWS, 64, D_INNER, nullptr, 0, nullptr);
        // 5) dt_proj + bias + softplus: (M,32)f32 x (1024,32)^T -> delta (M,1024)
        k_gemm_wmma<float, 8><<<(MROWS / 64) * (D_INNER / 128), 128, 0, stream>>>(
            dbcb, 64, dtWb + (size_t)l * D_INNER * DT_RANK,
            dly, D_INNER, MROWS, D_INNER, DT_RANK, dtB + l * D_INNER, 1, nullptr);
        // 6) selective scan + D-skip + SiLU(z) gate -> y (bf16)
        k_ssm_scan<<<(BATCH * D_INNER) / 256, 256, 0, stream>>>(
            dly, xcb, dbcb, xzb,
            A_log + (size_t)l * D_INNER * D_STATE, Dskip + l * D_INNER, ybf);
        // 7) out_proj + residual: (M,1024)bf16 x (512,1024)^T + xin -> xout
        k_gemm_wmma<bhalf, 8><<<(MROWS / 64) * (D_MODEL / 128), 128, 0, stream>>>(
            ybf, D_INNER, outWb + (size_t)l * D_MODEL * D_INNER,
            xout, D_MODEL, MROWS, D_MODEL, D_INNER, nullptr, 0, xin);
        xin = xout;
    }
    // 8) mean pool + classifier + sigmoid
    k_pool_cls<<<BATCH, 512, 0, stream>>>(xin, clsW, clsB, out);
}
